// TTNModel_11106785427997
// MI455X (gfx1250) — compile-verified, matrix operand order checked by reading, not run on hardware
//
#include <hip/hip_runtime.h>
#include <stdint.h>

typedef __attribute__((ext_vector_type(16))) __bf16 v16bf;
typedef __attribute__((ext_vector_type(8)))  float  v8f;

namespace {
constexpr int kWaves     = 4;          // 4 waves (wave32) per block = 128 threads
constexpr int kFragElems = 512;        // one 32x16 bf16 B-fragment: 32 lanes * 16 halves
// fragment region offsets inside d_ws, in __bf16 elements
constexpr int FR_A  = 0;                          // fused layers 7+6+5: 32 nodes * 8 chunks
constexpr int FR_L4 = FR_A  + 256 * kFragElems;   // 16 nodes * 8
constexpr int FR_L3 = FR_L4 + 128 * kFragElems;   //  8 nodes * 8
constexpr int FR_L2 = FR_L3 +  64 * kFragElems;   //  4 nodes * 8
constexpr int FR_L1 = FR_L2 +  32 * kFragElems;   //  2 nodes * 8
constexpr int FR_L0 = FR_L1 +  16 * kFragElems;   //  1 node  * 8 (N padded to 16)
} // namespace

union BF16x16 { v16bf v; __bf16 e[16]; unsigned u32[8]; uint4 q[2]; };

// packed bf16x2 multiply (VOP3P V_PK_MUL_BF16) -- keep bf16 math native
__device__ __forceinline__ unsigned pkmul_bf16(unsigned a, unsigned b) {
  unsigned d;
  asm("v_pk_mul_bf16 %0, %1, %2" : "=v"(d) : "v"(a), "v"(b));
  return d;
}
// pack two f32 into bf16x2
__device__ __forceinline__ unsigned cvt_pk_bf16(float lo, float hi) {
  unsigned d;
  asm("v_cvt_pk_bf16_f32 %0, %1, %2" : "=v"(d) : "v"(lo), "v"(hi));
  return d;
}
// splat one f32 into both bf16 halves (single cvt_pk)
__device__ __forceinline__ unsigned cvt_splat_bf16(float f) {
  unsigned d;
  asm("v_cvt_pk_bf16_f32 %0, %1, %2" : "=v"(d) : "v"(f), "v"(f));
  return d;
}
// splat low / high bf16 half across a 32-bit word (one v_perm_b32 each)
__device__ __forceinline__ unsigned splat_lo(unsigned x) {
  return __builtin_amdgcn_perm(x, x, 0x01000100u);
}
__device__ __forceinline__ unsigned splat_hi(unsigned x) {
  return __builtin_amdgcn_perm(x, x, 0x03020302u);
}

// ---------------------------------------------------------------------------
// Prep 1: fuse w7 (x4), w6 (x2), w5 into T[n5, p(8 leaf bits), k] and emit it
// directly in WMMA B-fragment layout (bf16).
// ---------------------------------------------------------------------------
__global__ void ttn_prep_fused(const float* __restrict__ w5,
                               const float* __restrict__ w6,
                               const float* __restrict__ w7,
                               __bf16* __restrict__ frags) {
  int tid = blockIdx.x * blockDim.x + threadIdx.x;
  if (tid >= 256 * kFragElems) return;
  int fragIdx = tid >> 9;            // node*8 + chunk
  int el      = tid & 511;
  int lane    = el >> 4;
  int e       = el & 15;
  int h       = lane >> 4;
  int k       = lane & 15;           // N (output index)
  int n       = fragIdx >> 3;        // layer-5 node
  int c       = fragIdx & 7;         // K-chunk
  // K/p index for (chunk, half, element) under the assumed 16-bit operand layout
  int p = ((e & 7) + 8 * h) + 16 * (e >> 3) + 32 * c;
  int b0 = p & 1, b1 = (p >> 1) & 1, b2 = (p >> 2) & 1, b3 = (p >> 3) & 1;
  int b4 = (p >> 4) & 1, b5 = (p >> 5) & 1, b6 = (p >> 6) & 1, b7 = (p >> 7) & 1;

  const float* w6L = w6 + (size_t)(2 * n)     * 256;   // [4,4,16]
  const float* w6R = w6 + (size_t)(2 * n + 1) * 256;
  const float* w7a = w7 + (((4 * n    ) * 2 + b0) * 2 + b1) * 4;   // [2,2,4] rows
  const float* w7b = w7 + (((4 * n + 1) * 2 + b2) * 2 + b3) * 4;
  const float* w7c = w7 + (((4 * n + 2) * 2 + b4) * 2 + b5) * 4;
  const float* w7d = w7 + (((4 * n + 3) * 2 + b6) * 2 + b7) * 4;

  float au[16], av[16];
  for (int u = 0; u < 16; ++u) {
    float sa = 0.f, sv = 0.f;
    for (int c0 = 0; c0 < 4; ++c0) {
      float ta = w7a[c0], tc = w7c[c0];
      for (int c1 = 0; c1 < 4; ++c1) {
        sa += w6L[(c0 * 4 + c1) * 16 + u] * ta * w7b[c1];
        sv += w6R[(c0 * 4 + c1) * 16 + u] * tc * w7d[c1];
      }
    }
    au[u] = sa; av[u] = sv;
  }
  const float* w5n = w5 + (size_t)n * 4096;            // [16,16,16]
  float val = 0.f;
  for (int u = 0; u < 16; ++u) {
    float tu = au[u];
    const float* w5r = w5n + (u * 16) * 16 + k;
    for (int v = 0; v < 16; ++v) val += tu * av[v] * w5r[v * 16];
  }
  frags[FR_A + fragIdx * kFragElems + lane * 16 + e] = (__bf16)val;
}

// ---------------------------------------------------------------------------
// Prep 2: shuffle w4..w0 into WMMA B-fragment layout (bf16). For the top
// tensor (N=2), pad N to 16 with zeros.
// ---------------------------------------------------------------------------
__global__ void ttn_prep_frags(const float* __restrict__ w4,
                               const float* __restrict__ w3,
                               const float* __restrict__ w2,
                               const float* __restrict__ w1,
                               const float* __restrict__ w0,
                               __bf16* __restrict__ frags) {
  int tid = blockIdx.x * blockDim.x + threadIdx.x;
  if (tid >= 248 * kFragElems) return;
  int fragIdx = tid >> 9;
  int el      = tid & 511;
  int lane    = el >> 4;
  int e       = el & 15;
  int h       = lane >> 4;
  int k       = lane & 15;

  const float* w; int rel; int base; bool top = false;
  if      (fragIdx < 128) { rel = fragIdx;       w = w4; base = FR_L4; }
  else if (fragIdx < 192) { rel = fragIdx - 128; w = w3; base = FR_L3; }
  else if (fragIdx < 224) { rel = fragIdx - 192; w = w2; base = FR_L2; }
  else if (fragIdx < 240) { rel = fragIdx - 224; w = w1; base = FR_L1; }
  else                    { rel = fragIdx - 240; w = w0; base = FR_L0; top = true; }
  int node = rel >> 3;
  int c    = rel & 7;
  int i = 2 * c + (e >> 3);          // K bits 4..7 (left-child index)
  int j = (e & 7) + 8 * h;           // K bits 0..3 (right-child index)
  float val;
  if (!top) val = w[(((size_t)node * 16 + i) * 16 + j) * 16 + k];
  else      val = (k < 2) ? w[(i * 16 + j) * 2 + k] : 0.0f;
  frags[base + rel * kFragElems + lane * 16 + e] = (__bf16)val;
}

// ---------------------------------------------------------------------------
// One 16x16x16 bilinear layer: NL nodes, in LDS -> out LDS.
// A[b,(i,j)] = u[i]*v[j] formed per lane in native bf16 (v_pk_mul_bf16);
// 8x v_wmma_f32_16x16x32_bf16 over K=256.
// ---------------------------------------------------------------------------
template <int NL>
__device__ __forceinline__ void ttn_layer_lds(const __bf16* __restrict__ frag,
                                              const __bf16* in, __bf16* outb,
                                              int lane) {
  const int m = lane & 15, h = lane >> 4;
  constexpr int sIn  = 2 * NL * 16;
  constexpr int sOut = NL * 16;
  for (int n = 0; n < NL; ++n) {
    const __bf16* fb = frag + n * 8 * kFragElems + lane * 16;
    __builtin_prefetch(fb + 8 * kFragElems, 0, 0);   // next node's fragments
    // u pairs: uu[c] = (u[2c], u[2c+1]) packed bf16x2
    uint4 ua = *reinterpret_cast<const uint4*>(in + m * sIn + (2 * n) * 16);
    uint4 ub = *reinterpret_cast<const uint4*>(in + m * sIn + (2 * n) * 16 + 8);
    unsigned uu[8] = {ua.x, ua.y, ua.z, ua.w, ub.x, ub.y, ub.z, ub.w};
    uint4 vq = *reinterpret_cast<const uint4*>(in + m * sIn + (2 * n + 1) * 16 + 8 * h);
    unsigned vp[4] = {vq.x, vq.y, vq.z, vq.w};
    v8f acc = {0.f, 0.f, 0.f, 0.f, 0.f, 0.f, 0.f, 0.f};
#pragma unroll
    for (int c = 0; c < 8; ++c) {
      unsigned slo = splat_lo(uu[c]);   // u[2c]   in both halves
      unsigned shi = splat_hi(uu[c]);   // u[2c+1] in both halves
      BF16x16 a, bm;
#pragma unroll
      for (int t = 0; t < 4; ++t) {
        a.u32[t]     = pkmul_bf16(slo, vp[t]);
        a.u32[t + 4] = pkmul_bf16(shi, vp[t]);
      }
      const uint4* bq = reinterpret_cast<const uint4*>(fb + c * kFragElems);
      bm.q[0] = bq[0]; bm.q[1] = bq[1];
      acc = __builtin_amdgcn_wmma_f32_16x16x32_bf16(false, a.v, false, bm.v,
                                                    (short)0, acc, false, false);
    }
    __bf16* o = outb + n * 16 + m;   // D: lane holds column N=m, rows r+8h
#pragma unroll
    for (int r = 0; r < 8; ++r) o[(r + 8 * h) * sOut] = (__bf16)acc[r];
  }
}

// Top layer: 1 node, only k=0,1 valid; write f32 output to global.
__device__ __forceinline__ void ttn_layer_final(const __bf16* __restrict__ frag,
                                                const __bf16* in,
                                                float* __restrict__ out,
                                                int lane, int tileBase) {
  const int m = lane & 15, h = lane >> 4;
  uint4 ua = *reinterpret_cast<const uint4*>(in + m * 32);
  uint4 ub = *reinterpret_cast<const uint4*>(in + m * 32 + 8);
  unsigned uu[8] = {ua.x, ua.y, ua.z, ua.w, ub.x, ub.y, ub.z, ub.w};
  uint4 vq = *reinterpret_cast<const uint4*>(in + m * 32 + 16 + 8 * h);
  unsigned vp[4] = {vq.x, vq.y, vq.z, vq.w};
  v8f acc = {0.f, 0.f, 0.f, 0.f, 0.f, 0.f, 0.f, 0.f};
  const __bf16* fb = frag + lane * 16;
#pragma unroll
  for (int c = 0; c < 8; ++c) {
    unsigned slo = splat_lo(uu[c]);
    unsigned shi = splat_hi(uu[c]);
    BF16x16 a, bm;
#pragma unroll
    for (int t = 0; t < 4; ++t) {
      a.u32[t]     = pkmul_bf16(slo, vp[t]);
      a.u32[t + 4] = pkmul_bf16(shi, vp[t]);
    }
    const uint4* bq = reinterpret_cast<const uint4*>(fb + c * kFragElems);
    bm.q[0] = bq[0]; bm.q[1] = bq[1];
    acc = __builtin_amdgcn_wmma_f32_16x16x32_bf16(false, a.v, false, bm.v,
                                                  (short)0, acc, false, false);
  }
  if (m < 2) {
#pragma unroll
    for (int r = 0; r < 8; ++r)
      out[(size_t)(tileBase + r + 8 * h) * 2 + m] = acc[r];
  }
}

// ---------------------------------------------------------------------------
// Main kernel: each wave carries a 16-sample tile through all layers.
// LDS per wave: r0 = 16 samples * 512 bf16 (16KB), r1 = 16 * 256 bf16 (8KB).
// ---------------------------------------------------------------------------
__global__ __launch_bounds__(kWaves * 32)
void ttn_main(const float* __restrict__ x,
              const __bf16* __restrict__ frags,
              float* __restrict__ out) {
  extern __shared__ __bf16 smem[];
  const int lane = threadIdx.x & 31;
  const int wv   = threadIdx.x >> 5;
  const int m    = lane & 15;
  const int h    = lane >> 4;
  const int tileBase = (blockIdx.x * kWaves + wv) * 16;
  __bf16* r0 = smem + wv * 12288;
  __bf16* r1 = r0 + 8192;

  // ---- Stage A: fused layers 7+6+5. A = 8-leaf outer product, K = 256. ----
  {
    const int b = tileBase + m;
    const float4* xq = reinterpret_cast<const float4*>(x + (size_t)b * 512);
    for (int nd = 0; nd < 32; ++nd) {
      const __bf16* fb = frags + FR_A + nd * 8 * kFragElems + lane * 16;
      __builtin_prefetch(fb + 8 * kFragElems, 0, 0);
      float4 q0 = xq[nd * 4 + 0], q1 = xq[nd * 4 + 1];
      float4 q2 = xq[nd * 4 + 2], q3 = xq[nd * 4 + 3];
      // leaf t values: l{t}a = x[leaf t][0], l{t}b = x[leaf t][1]
      float l0a = q0.x, l0b = q0.y, l1a = q0.z, l1b = q0.w;
      float l2a = q1.x, l2b = q1.y, l3a = q1.z, l3b = q1.w;
      float l4a = q2.x, l4b = q2.y, l5a = q2.z, l5b = q2.w;
      float l6a = q3.x, l6b = q3.y, l7a = q3.z, l7b = q3.w;
      float g01[4] = { l0a * l1a, l0b * l1a, l0a * l1b, l0b * l1b };
      float g012[8];
#pragma unroll
      for (int t = 0; t < 8; ++t) g012[t] = g01[t & 3] * ((t & 4) ? l2b : l2a);
      // per-node bf16 table t4[e] = g012[e&7] * x4[e>>3]
      unsigned t4p[8];
#pragma unroll
      for (int t = 0; t < 4; ++t) {
        t4p[t]     = cvt_pk_bf16(g012[2 * t] * l4a, g012[2 * t + 1] * l4a);
        t4p[t + 4] = cvt_pk_bf16(g012[2 * t] * l4b, g012[2 * t + 1] * l4b);
      }
      float x3h = h ? l3b : l3a;
      v8f acc = {0.f, 0.f, 0.f, 0.f, 0.f, 0.f, 0.f, 0.f};
#pragma unroll
      for (int c = 0; c < 8; ++c) {
        float F  = ((c & 1) ? l5b : l5a) * ((c & 2) ? l6b : l6a) * ((c & 4) ? l7b : l7a);
        unsigned bs2 = cvt_splat_bf16(x3h * F);   // chunk scalar, splat bf16x2
        BF16x16 a, bm;
#pragma unroll
        for (int t = 0; t < 8; ++t) a.u32[t] = pkmul_bf16(t4p[t], bs2);
        const uint4* bq = reinterpret_cast<const uint4*>(fb + c * kFragElems);
        bm.q[0] = bq[0]; bm.q[1] = bq[1];
        acc = __builtin_amdgcn_wmma_f32_16x16x32_bf16(false, a.v, false, bm.v,
                                                      (short)0, acc, false, false);
      }
      __bf16* o = r0 + nd * 16 + m;
#pragma unroll
      for (int r = 0; r < 8; ++r) o[(r + 8 * h) * 512] = (__bf16)acc[r];
    }
  }
  __syncthreads();
  ttn_layer_lds<16>(frags + FR_L4, r0, r1, lane);   // 32 vecs -> 16
  __syncthreads();
  ttn_layer_lds<8>(frags + FR_L3, r1, r0, lane);    // 16 -> 8
  __syncthreads();
  ttn_layer_lds<4>(frags + FR_L2, r0, r1, lane);    // 8 -> 4
  __syncthreads();
  ttn_layer_lds<2>(frags + FR_L1, r1, r0, lane);    // 4 -> 2
  __syncthreads();
  ttn_layer_final(frags + FR_L0, r0, out, lane, tileBase);  // 2 -> [B,2]
}

extern "C" void kernel_launch(void* const* d_in, const int* in_sizes, int n_in,
                              void* d_out, int out_size, void* d_ws, size_t ws_size,
                              hipStream_t stream) {
  (void)in_sizes; (void)n_in; (void)out_size; (void)ws_size;
  const float* x  = (const float*)d_in[0];
  const float* w0 = (const float*)d_in[1];   // (1,16,16,2)
  const float* w1 = (const float*)d_in[2];   // (2,16,16,16)
  const float* w2 = (const float*)d_in[3];   // (4,16,16,16)
  const float* w3 = (const float*)d_in[4];   // (8,16,16,16)
  const float* w4 = (const float*)d_in[5];   // (16,16,16,16)
  const float* w5 = (const float*)d_in[6];   // (32,16,16,16)
  const float* w6 = (const float*)d_in[7];   // (64,4,4,16)
  const float* w7 = (const float*)d_in[8];   // (128,2,2,4)
  __bf16* frags = (__bf16*)d_ws;             // 504 fragments * 1KB = 504KB

  ttn_prep_fused<<<512, 256, 0, stream>>>(w5, w6, w7, frags);
  ttn_prep_frags<<<496, 256, 0, stream>>>(w4, w3, w2, w1, w0, frags);

  const int blocks = 32768 / (kWaves * 16);  // 512 blocks of 4 waves
  const size_t lds = (size_t)kWaves * 12288 * sizeof(__bf16);  // 96KB
  ttn_main<<<blocks, kWaves * 32, lds, stream>>>(x, frags, (float*)d_out);
}
